// TriangleAttention_20160576487725
// MI455X (gfx1250) — compile-verified, matrix operand order checked by reading, not run on hardware
//
#include <hip/hip_runtime.h>

// Triangle attention (AlphaFold-style), B=1, N=256, D=128, H=4, E=32.
// k_prep:  one-time weight transpose/convert to f16 (Wcat, WoT) in workspace.
// k_qkvb:  fused QKV+bias projection GEMM via WMMA.
// k_attn:  per-(h,i) attention (S=QK^T, softmax, O=PV) via WMMA.
// k_out:   output projection GEMM via WMMA, f32 result.

#define N_TOK 256
#define D_MOD 128
#define N_HEAD 4
#define E_DIM 32
#define P_TOT (N_TOK * N_TOK)

typedef __attribute__((ext_vector_type(16))) _Float16 v16h;
typedef __attribute__((ext_vector_type(4)))  _Float16 v4h;
typedef __attribute__((ext_vector_type(8)))  float    v8f;

// Load a 16x32 f16 WMMA A/B operand from LDS.
// Logical element (m, kk) lives at base[m*stride + kk] (f16 units).
// CDNA5 16-bit operand layout (CDNA5 ISA):
//   lanes  0-15: m = lane,     VGPR0..3 = K{0..7},  VGPR4..7 = K{16..23}
//   lanes 16-31: m = lane-16,  VGPR0..3 = K{8..15}, VGPR4..7 = K{24..31}
// K pairs are dword-packed -> two 16B-aligned dword-quad LDS loads.
static __device__ __forceinline__ v16h load_op16(const _Float16* base, int stride, int lane) {
  const int m = lane & 15;
  const int half = (lane >> 4) & 1;
  const unsigned* p = (const unsigned*)(base + m * stride);
  union { v16h v; unsigned u[8]; } r;
#pragma unroll
  for (int t = 0; t < 4; ++t) r.u[t] = p[(half << 2) + t];
#pragma unroll
  for (int t = 0; t < 4; ++t) r.u[4 + t] = p[8 + (half << 2) + t];
  return r.v;
}

// One 16x16 C tile of A[16x128] x B[128x16]; all 4 B operands fetched up front
// so the 4 WMMAs issue back-to-back.
static __device__ __forceinline__ v8f mm_tile128(const _Float16* Wc, const v16h (&a)[4],
                                                 int nt, int lane) {
  v16h b[4];
#pragma unroll
  for (int ks = 0; ks < 4; ++ks)
    b[ks] = load_op16(Wc + nt * 16 * 128 + ks * 32, 128, lane);
  v8f acc = {};
#pragma unroll
  for (int ks = 0; ks < 4; ++ks)
    acc = __builtin_amdgcn_wmma_f32_16x16x32_f16(false, a[ks], false, b[ks],
                                                 (short)0, acc, false, false);
  return acc;
}

// Stage a 16x16 f16 C tile in a per-wave LDS strip, then write it out as one
// 16B global store per lane (2 lanes per row). gbase = tile origin, 16 cols
// contiguous, row stride in f16 units. All addresses 16B-aligned.
static __device__ __forceinline__ void store_tile_f16(_Float16* gbase, int row_stride,
                                                      _Float16* stage, const v8f& acc,
                                                      int lane) {
  const int half = (lane >> 4) & 1, ncol = lane & 15;
#pragma unroll
  for (int r = 0; r < 8; ++r)
    stage[((r + (half << 3)) << 4) + ncol] = (_Float16)acc[r];
  const int row = lane >> 1, seg = lane & 1;
  const uint4 d = *(const uint4*)(stage + (row << 4) + (seg << 3));
  *(uint4*)(gbase + row * row_stride + (seg << 3)) = d;
}

// Same idea for a 16x16 f32 tile: two float4 stores per lane.
static __device__ __forceinline__ void store_tile_f32(float* gbase, int row_stride,
                                                      float* stage, const v8f& acc,
                                                      int lane) {
  const int half = (lane >> 4) & 1, ncol = lane & 15;
#pragma unroll
  for (int r = 0; r < 8; ++r)
    stage[((r + (half << 3)) << 4) + ncol] = acc[r];
  const int row = lane >> 1, seg = lane & 1;
  const float4* sp = (const float4*)(stage + (row << 4) + (seg << 3));
  float4* gp = (float4*)(gbase + row * row_stride + (seg << 3));
  gp[0] = sp[0];
  gp[1] = sp[1];
}

// ---------------------------------------------------------------------------
// Kernel 0: one block. Build f16 transposed weights in workspace:
//   Wcat[400][128]: rows 0..127 Wq^T * scale, 128..255 Wk^T, 256..383 Wv^T,
//                   384..387 Wb^T, 388..399 zero.   WoT[128][128] = Wo^T.
// ---------------------------------------------------------------------------
__global__ __launch_bounds__(256) void k_prep(
    const float* __restrict__ Wq, const float* __restrict__ Wk,
    const float* __restrict__ Wv, const float* __restrict__ Wb,
    const float* __restrict__ Wo,
    _Float16* __restrict__ Wcat_g, _Float16* __restrict__ WoT_g)
{
  const int tid = threadIdx.x;
  const float scale = 0.17677669529663687f;  // E^-0.5, folded into Wq
  for (int idx = tid; idx < 400 * 128; idx += 256) {
    const int c = idx >> 7, kk = idx & 127;
    float w;
    if (c < 128)      w = Wq[kk * 128 + c] * scale;
    else if (c < 256) w = Wk[kk * 128 + (c - 128)];
    else if (c < 384) w = Wv[kk * 128 + (c - 256)];
    else if (c < 388) w = Wb[kk * 4 + (c - 384)];
    else              w = 0.0f;
    Wcat_g[idx] = (_Float16)w;
  }
  for (int idx = tid; idx < 128 * 128; idx += 256) {
    const int c = idx >> 7, kk = idx & 127;
    WoT_g[idx] = (_Float16)Wo[kk * 128 + c];
  }
}

// ---------------------------------------------------------------------------
// Kernel 1: fused QKV + bias projection.
// C = [65536 x 400] = pair[65536 x 128] x Wcat^T. 512 blocks x 8 waves.
// ---------------------------------------------------------------------------
__global__ __launch_bounds__(256) void k_qkvb(
    const float* __restrict__ pair, const _Float16* __restrict__ Wcat_g,
    _Float16* __restrict__ qb, _Float16* __restrict__ kb,
    _Float16* __restrict__ vb, float* __restrict__ biasb)
{
  __shared__ _Float16 Apair[128 * 128];   // 32 KB
  __shared__ _Float16 Wcat[400 * 128];    // 100 KB
  __shared__ _Float16 Cstage[8 * 256];    // 4 KB (per-wave tile strips)
  const int tid = threadIdx.x;
  const int p0 = blockIdx.x * 128;

  {  // 16B-vector LDS fills
    const uint4* src = (const uint4*)Wcat_g;
    uint4* dst = (uint4*)Wcat;
    for (int idx = tid; idx < (400 * 128) / 8; idx += 256) dst[idx] = src[idx];
  }
  {
    const float4* src = (const float4*)(pair + ((size_t)p0 << 7));
    v4h* dst = (v4h*)Apair;
    for (int idx = tid; idx < (128 * 128) / 4; idx += 256) {
      const float4 f = src[idx];
      v4h h4 = {(_Float16)f.x, (_Float16)f.y, (_Float16)f.z, (_Float16)f.w};
      dst[idx] = h4;
    }
  }
  __syncthreads();

  const int wave = tid >> 5, lane = tid & 31;
  const int half = (lane >> 4) & 1;
  const int mtile = wave;
  _Float16* stg = Cstage + wave * 256;

  v16h a[4];
#pragma unroll
  for (int ks = 0; ks < 4; ++ks)
    a[ks] = load_op16(Apair + mtile * 16 * 128 + ks * 32, 128, lane);

  // q / k / v regions: each N-tile maps entirely to one destination tensor,
  // and within a tile the head index and e-base are uniform.
  for (int which = 0; which < 3; ++which) {
    _Float16* dst = (which == 0) ? qb : (which == 1) ? kb : vb;
    for (int ntl = 0; ntl < 8; ++ntl) {
      const v8f acc = mm_tile128(Wcat, a, which * 8 + ntl, lane);
      const int h = ntl >> 1;               // head (uniform per tile)
      const int ebase = (ntl & 1) << 4;     // 0 or 16 (uniform per tile)
      _Float16* gbase = dst + ((size_t)h << 21) +
                        ((size_t)(p0 + mtile * 16) << 5) + ebase;
      store_tile_f16(gbase, 32, stg, acc, lane);
    }
  }
  // bias tile (nt = 24): cols 0..3 are Wb, rest zero-padded.
  {
    const v8f acc = mm_tile128(Wcat, a, 24, lane);
    if ((lane & 15) < 4) {
#pragma unroll
      for (int r = 0; r < 8; ++r) {
        const size_t p = p0 + mtile * 16 + r + (half << 3);
        biasb[(size_t)(lane & 15) * P_TOT + p] = acc[r];
      }
    }
  }
}

// ---------------------------------------------------------------------------
// Kernel 2: attention, one block per (h, i). 256 threads = 8 waves,
// each wave handles two 16-row j-tiles sequentially.
// ---------------------------------------------------------------------------
__global__ __launch_bounds__(256) void k_attn(
    const _Float16* __restrict__ qb, const _Float16* __restrict__ kb,
    const _Float16* __restrict__ vb, const float* __restrict__ biasb,
    const unsigned char* __restrict__ mask, _Float16* __restrict__ ob)
{
  __shared__ _Float16 Qs[256 * 32];       // 16 KB  [j][e]
  __shared__ _Float16 Ks[256 * 32];       // 16 KB  [k][e]
  __shared__ _Float16 Vt[32 * 256];       // 16 KB  [e][k] (transposed)
  __shared__ _Float16 Pb[8 * 16 * 256];   // 64 KB  per-wave P strips
  __shared__ _Float16 Ostage[8 * 256];    // 4 KB   per-wave store strips
  __shared__ float    maskadd[256];       // 1 KB
  const int tid = threadIdx.x;
  const int h = blockIdx.x >> 8;
  const int i = blockIdx.x & 255;
  const size_t base = ((size_t)h << 21) + ((size_t)i << 13);

  {  // vectorized Q/K fills
    const uint4* qs = (const uint4*)(qb + base);
    const uint4* ks = (const uint4*)(kb + base);
    uint4* qd = (uint4*)Qs;
    uint4* kd = (uint4*)Ks;
    for (int idx = tid; idx < (256 * 32) / 8; idx += 256) {
      qd[idx] = qs[idx];
      kd[idx] = ks[idx];
    }
  }
  {  // V transposed into LDS
    const _Float16* vs = vb + base;
    for (int idx = tid; idx < 256 * 32; idx += 256) {
      const int j = idx >> 5, e = idx & 31;
      Vt[e * 256 + j] = vs[idx];
    }
  }
  maskadd[tid] = mask[tid] ? 0.0f : -1e30f;
  __syncthreads();

  const int wave = tid >> 5, lane = tid & 31;
  const int half = (lane >> 4) & 1, ncol = lane & 15;
  _Float16* Pw = Pb + wave * 16 * 256;
  _Float16* stg = Ostage + wave * 256;
  const float* brow = biasb + ((size_t)h << 16);

  for (int t = 0; t < 2; ++t) {
    const int jt = wave + t * 8;

    // S = Q K^T (scale already folded into q); B operands in groups of 4.
    const v16h aq = load_op16(Qs + jt * 16 * 32, 32, lane);
    v8f s[16];
#pragma unroll
    for (int kg = 0; kg < 4; ++kg) {
      v16h b[4];
#pragma unroll
      for (int u = 0; u < 4; ++u)
        b[u] = load_op16(Ks + (kg * 4 + u) * 16 * 32, 32, lane);
#pragma unroll
      for (int u = 0; u < 4; ++u) {
        v8f z = {};
        s[kg * 4 + u] = __builtin_amdgcn_wmma_f32_16x16x32_f16(
            false, aq, false, b[u], (short)0, z, false, false);
      }
    }

    // + bias[h, j, k] + mask(k)
#pragma unroll
    for (int kt = 0; kt < 16; ++kt) {
      const int kk = kt * 16 + ncol;
      const float ma = maskadd[kk];
#pragma unroll
      for (int r = 0; r < 8; ++r) {
        const int jj = jt * 16 + r + (half << 3);
        s[kt][r] = s[kt][r] + brow[((size_t)jj << 8) + kk] + ma;
      }
    }

    // row softmax: row r lives in VGPR slot r across the 16 lanes of a half.
    float rowmax[8], rowsum[8];
#pragma unroll
    for (int r = 0; r < 8; ++r) {
      float m = s[0][r];
#pragma unroll
      for (int kt = 1; kt < 16; ++kt) m = fmaxf(m, s[kt][r]);
#pragma unroll
      for (int off = 8; off >= 1; off >>= 1) m = fmaxf(m, __shfl_xor(m, off, 32));
      rowmax[r] = m;
    }
#pragma unroll
    for (int r = 0; r < 8; ++r) {
      float sum = 0.0f;
#pragma unroll
      for (int kt = 0; kt < 16; ++kt) {
        const float e = __expf(s[kt][r] - rowmax[r]);
        s[kt][r] = e;
        sum += e;
      }
#pragma unroll
      for (int off = 8; off >= 1; off >>= 1) sum += __shfl_xor(sum, off, 32);
      rowsum[r] = 1.0f / sum;
    }

    // normalize, write P (f16) to this wave's LDS strip
#pragma unroll
    for (int kt = 0; kt < 16; ++kt) {
      const int kk = kt * 16 + ncol;
#pragma unroll
      for (int r = 0; r < 8; ++r) {
        const int lr = r + (half << 3);
        Pw[lr * 256 + kk] = (_Float16)(s[kt][r] * rowsum[r]);
      }
    }

    // O = P * V: ks outer so each P operand feeds both e-tiles (2 WMMAs/step).
    v8f acc0 = {}, acc1 = {};
#pragma unroll
    for (int ks = 0; ks < 8; ++ks) {
      const v16h pa = load_op16(Pw + ks * 32, 256, lane);
      const v16h v0 = load_op16(Vt + ks * 32, 256, lane);
      const v16h v1 = load_op16(Vt + 16 * 256 + ks * 32, 256, lane);
      acc0 = __builtin_amdgcn_wmma_f32_16x16x32_f16(false, pa, false, v0,
                                                    (short)0, acc0, false, false);
      acc1 = __builtin_amdgcn_wmma_f32_16x16x32_f16(false, pa, false, v1,
                                                    (short)0, acc1, false, false);
    }
    _Float16* gbase = ob + (((size_t)i * N_TOK + jt * 16) << 7) + h * E_DIM;
    store_tile_f16(gbase, 128, stg, acc0, lane);
    store_tile_f16(gbase + 16, 128, stg, acc1, lane);
  }
}

// ---------------------------------------------------------------------------
// Kernel 3: out = [65536 x 128] x Wo[128 x 128], f32 output.
// ---------------------------------------------------------------------------
__global__ __launch_bounds__(256) void k_out(
    const _Float16* __restrict__ ob, const _Float16* __restrict__ WoT_g,
    float* __restrict__ out)
{
  __shared__ _Float16 As[128 * 128];  // 32 KB
  __shared__ _Float16 Wt[128 * 128];  // 32 KB
  __shared__ float    Fstage[8 * 256];  // 8 KB
  const int tid = threadIdx.x;
  const size_t p0 = (size_t)blockIdx.x * 128;

  {
    const uint4* asrc = (const uint4*)(ob + (p0 << 7));
    const uint4* wsrc = (const uint4*)WoT_g;
    uint4* adst = (uint4*)As;
    uint4* wdst = (uint4*)Wt;
    for (int idx = tid; idx < (128 * 128) / 8; idx += 256) {
      adst[idx] = asrc[idx];
      wdst[idx] = wsrc[idx];
    }
  }
  __syncthreads();

  const int wave = tid >> 5, lane = tid & 31;
  const int mtile = wave;
  float* stg = Fstage + wave * 256;

  v16h a[4];
#pragma unroll
  for (int ks = 0; ks < 4; ++ks)
    a[ks] = load_op16(As + mtile * 16 * 128 + ks * 32, 128, lane);

  for (int nt = 0; nt < 8; ++nt) {
    const v8f acc = mm_tile128(Wt, a, nt, lane);
    float* gbase = out + ((p0 + mtile * 16) << 7) + nt * 16;
    store_tile_f32(gbase, 128, stg, acc, lane);
  }
}

// ---------------------------------------------------------------------------
// Launch. Inputs: pairwise_repr, mask, Wb, Wq, Wk, Wv, Wo.
// Workspace: q,k,v,o (f16, 16 MB each) | bias (f32, 1 MB) | Wcat, WoT (f16).
// ---------------------------------------------------------------------------
extern "C" void kernel_launch(void* const* d_in, const int* in_sizes, int n_in,
                              void* d_out, int out_size, void* d_ws, size_t ws_size,
                              hipStream_t stream) {
  const float*         pair = (const float*)d_in[0];
  const unsigned char* mask = (const unsigned char*)d_in[1];
  const float*         Wb   = (const float*)d_in[2];
  const float*         Wq   = (const float*)d_in[3];
  const float*         Wk   = (const float*)d_in[4];
  const float*         Wv   = (const float*)d_in[5];
  const float*         Wo   = (const float*)d_in[6];

  const size_t nHE = (size_t)N_HEAD * N_TOK * N_TOK * E_DIM;  // 8M elems
  _Float16* qb     = (_Float16*)d_ws;
  _Float16* kb     = qb + nHE;
  _Float16* vb     = kb + nHE;
  _Float16* ob     = vb + nHE;
  float*    biasb  = (float*)(ob + nHE);
  _Float16* Wcat_g = (_Float16*)(biasb + (size_t)N_HEAD * P_TOT);
  _Float16* WoT_g  = Wcat_g + 400 * 128;

  k_prep<<<1, 256, 0, stream>>>(Wq, Wk, Wv, Wb, Wo, Wcat_g, WoT_g);
  k_qkvb<<<P_TOT / 128, 256, 0, stream>>>(pair, Wcat_g, qb, kb, vb, biasb);
  k_attn<<<N_HEAD * N_TOK, 256, 0, stream>>>(qb, kb, vb, biasb, mask, ob);
  k_out<<<P_TOT / 128, 256, 0, stream>>>(ob, WoT_g, (float*)d_out);
}
